// DeformableAttention1D_51470888075694
// MI455X (gfx1250) — matrix operand order repaired and checked by reading, not compiled, not run on hardware
//
#include <hip/hip_runtime.h>
#include <hip/hip_bf16.h>
#include <stdint.h>

// ---------------------------------------------------------------------------
// DeformableAttention1D for MI455X (gfx1250, wave32, WMMA).
// Attention core: v_wmma_f32_16x16x32_f16 with double-buffered async-to-LDS
// K/V staging (ASYNCcnt), online softmax with WMMA-accumulated row sums and
// v_permlane16 row-max reductions. Everything else is fp32 bandwidth glue.
// ---------------------------------------------------------------------------

typedef __attribute__((ext_vector_type(16))) _Float16 v16h;
typedef __attribute__((ext_vector_type(8)))  _Float16 v8h;
typedef __attribute__((ext_vector_type(8)))  float    v8f;

union H16 { v16h v; v8h h[2]; };

#define N_SEQ   4096
#define M_SEQ   1024
#define DIMC    512
#define DHEAD   64
#define KSZ     6
#define DSF     4
#define SCALE_Q 0.125f          // 64^-0.5
#define OFF_SC  4.0f
#define LS      72              // LDS row stride (halves): 144B rows, 16B aligned,
                                // row*36 mod 64 banks distinct for 16 rows

// --- gfx1250 async global->LDS copy (ASYNCcnt path). LDS operand is the
// wave-relative LDS byte address (low 32 bits of the flat shared pointer).
__device__ __forceinline__ void async_load_b128(void* lds, const void* g) {
#if defined(__gfx1250__)
  asm volatile("global_load_async_to_lds_b128 %0, %1, off"
               :: "v"((uint32_t)(uintptr_t)lds),
                  "v"((unsigned long long)(uintptr_t)g)
               : "memory");
#else
  ((int4*)lds)[0] = ((const int4*)g)[0];
#endif
}
__device__ __forceinline__ void wait_async0() {
#if defined(__gfx1250__)
  asm volatile("s_wait_asynccnt 0" ::: "memory");
#endif
}
// Each wave issues exactly 4 async ops per staged buffer; async loads complete
// in order, so waiting for <=4 outstanding retires the previous buffer while
// the freshly issued one stays in flight (double buffering).
__device__ __forceinline__ void wait_async4() {
#if defined(__gfx1250__)
  asm volatile("s_wait_asynccnt 4" ::: "memory");
#endif
}

// --- VALU cross-lane reductions within each 16-lane half (v_permlane16_b32,
// xor-butterfly lane selects as immediates) -- keeps the LDS pipe free.
__device__ __forceinline__ float pl16(float v, uint32_t s0, uint32_t s1) {
#if defined(__gfx1250__)
  return __uint_as_float(__builtin_amdgcn_permlane16(
      __float_as_uint(v), __float_as_uint(v), s0, s1, false, false));
#else
  return v;
#endif
}
__device__ __forceinline__ float rowmax16(float v) {
  v = fmaxf(v, pl16(v, 0x67452301u, 0xEFCDAB89u));   // xor 1
  v = fmaxf(v, pl16(v, 0x54761032u, 0xDCFE98BAu));   // xor 2
  v = fmaxf(v, pl16(v, 0x32107654u, 0xBA98FEDCu));   // xor 4
  v = fmaxf(v, pl16(v, 0xFEDCBA98u, 0x76543210u));   // xor 8
  return v;
}
__device__ __forceinline__ float bcast16_lane0(float v) {
  return pl16(v, 0x00000000u, 0x00000000u);          // all lanes read lane 0
}

// ---------------------------------------------------------------------------
// Kernel 1: grouped 1x1 conv q = Wq * concat(prev_x, x)   (group g<4 reads
// prev_x channels g*128.., g>=4 reads x channels (g-4)*128..).
// Emits q fp32 in [c][t] (for offset net) and q*SCALE f16 in [h][t][d].
// ---------------------------------------------------------------------------
__global__ __launch_bounds__(256) void qproj_kernel(
    const float* __restrict__ x, const float* __restrict__ prev_x,
    const float* __restrict__ Wq, float* __restrict__ q,
    _Float16* __restrict__ qsf16)
{
  __shared__ float xin[32 * 129];       // padded: bank = (tt+ic)%64
  __shared__ float wq[64 * 128];
  const int g   = blockIdx.x;           // 0..7
  const int t0  = blockIdx.y * 32;
  const int tid = threadIdx.x;
  const float* src  = (g < 4) ? prev_x : x;
  const int    cb   = (g < 4) ? g * 128 : (g - 4) * 128;

  for (int i = tid; i < 64 * 128; i += 256) wq[i] = Wq[g * 64 * 128 + i];
  for (int i = tid; i < 32 * 128; i += 256) {
    int tt = i >> 7, ic = i & 127;
    xin[tt * 129 + ic] = src[(size_t)(t0 + tt) * DIMC + cb + ic];
  }
  __syncthreads();

  for (int i = tid; i < 64 * 32; i += 256) {
    int oc = i >> 5, tt = i & 31;       // consecutive tid -> consecutive t
    float acc = 0.f;
    #pragma unroll 4
    for (int ic = 0; ic < 128; ++ic)
      acc += wq[oc * 128 + ic] * xin[tt * 129 + ic];
    int t = t0 + tt;
    q[(size_t)(g * 64 + oc) * N_SEQ + t] = acc;
    qsf16[((size_t)g * N_SEQ + t) * DHEAD + oc] = (_Float16)(acc * SCALE_Q);
  }
}

// ---------------------------------------------------------------------------
// Kernel 2: offset network. One wave per (g, j): depthwise conv (K=6, stride 4,
// pad 1) -> exact GeLU -> 1x1 to scalar -> tanh*4 -> grid_sample coords.
// ---------------------------------------------------------------------------
__global__ __launch_bounds__(256) void offset_kernel(
    const float* __restrict__ q, const float* __restrict__ Wdw,
    const float* __restrict__ bdw, const float* __restrict__ Wp,
    int* __restrict__ sidx, float* __restrict__ sw)
{
  const int tid  = threadIdx.x;
  const int wave = tid >> 5, lane = tid & 31;
  const int widx = blockIdx.x * 8 + wave;       // 0..8191
  const int g = widx >> 10, j = widx & 1023;

  float acc = 0.f;
  #pragma unroll
  for (int half = 0; half < 2; ++half) {
    int c = lane + half * 32;                   // each lane: 2 channels
    float h = bdw[c];
    #pragma unroll
    for (int k = 0; k < KSZ; ++k) {
      int tq = j * DSF - 1 + k;
      if (tq >= 0 && tq < N_SEQ)
        h += Wdw[c * KSZ + k] * q[(size_t)(g * DHEAD + c) * N_SEQ + tq];
    }
    float ge = 0.5f * h * (1.f + erff(h * 0.70710678118654752f));  // exact gelu
    acc += Wp[c] * ge;
  }
  #pragma unroll
  for (int d = 16; d >= 1; d >>= 1) acc += __shfl_xor(acc, d, 32);

  if (lane == 0) {
    float off = tanhf(acc) * OFF_SC;
    // pos = (j+off)*N/(m-1) - 0.5  (align_corners=False unnormalize)
    float pos = ((float)j + off) * ((float)N_SEQ / (float)(M_SEQ - 1)) - 0.5f;
    float x0  = floorf(pos);
    sidx[g * M_SEQ + j] = (int)x0;
    sw[g * M_SEQ + j]   = pos - x0;
  }
}

// ---------------------------------------------------------------------------
// Kernel 3: bilinear grid-sample (zero padding) + grouped 1x1 convs K,V.
// Emits f16 K[bh][j][d] (row-major j) and f16 V^T[bh][d][j].
// ---------------------------------------------------------------------------
__global__ __launch_bounds__(256) void kv_kernel(
    const float* __restrict__ x, const float* __restrict__ prev_x,
    const float* __restrict__ Wk, const float* __restrict__ Wv,
    const int* __restrict__ sidx, const float* __restrict__ sw,
    _Float16* __restrict__ kf16, _Float16* __restrict__ vf16T)
{
  __shared__ float kvs[32 * 65];
  __shared__ float wk[64 * 65];
  __shared__ float wv[64 * 65];
  const int bb  = blockIdx.x;                   // b2*8 + g
  const int j0  = blockIdx.y * 32;
  const int g   = bb & 7, b2 = bb >> 3;
  const int tid = threadIdx.x;
  const float* src = (b2 == 0) ? prev_x : x;

  for (int i = tid; i < 64 * 64; i += 256) {
    int oc = i >> 6, ic = i & 63;
    wk[oc * 65 + ic] = Wk[(size_t)(g * 64 + oc) * 64 + ic];
    wv[oc * 65 + ic] = Wv[(size_t)(g * 64 + oc) * 64 + ic];
  }
  for (int i = tid; i < 32 * 64; i += 256) {
    int jj = i >> 6, ic = i & 63;
    int   ti = sidx[g * M_SEQ + j0 + jj];
    float w1 = sw[g * M_SEQ + j0 + jj];
    int c = g * 64 + ic;
    float a = (ti     >= 0 && ti     < N_SEQ) ? src[(size_t)ti * DIMC + c]       : 0.f;
    float b = (ti + 1 >= 0 && ti + 1 < N_SEQ) ? src[(size_t)(ti + 1) * DIMC + c] : 0.f;
    kvs[jj * 65 + ic] = a * (1.f - w1) + b * w1;
  }
  __syncthreads();

  for (int i = tid; i < 32 * 64; i += 256) {    // K: oc fastest -> coalesced
    int jj = i >> 6, oc = i & 63;
    float acc = 0.f;
    #pragma unroll 4
    for (int ic = 0; ic < 64; ++ic) acc += wk[oc * 65 + ic] * kvs[jj * 65 + ic];
    kf16[((size_t)bb * M_SEQ + j0 + jj) * 64 + oc] = (_Float16)acc;
  }
  for (int i = tid; i < 32 * 64; i += 256) {    // V^T: jj fastest -> coalesced
    int oc = i >> 5, jj = i & 31;
    float acc = 0.f;
    #pragma unroll 4
    for (int ic = 0; ic < 64; ++ic) acc += wv[oc * 65 + ic] * kvs[jj * 65 + ic];
    vf16T[((size_t)bb * 64 + oc) * M_SEQ + j0 + jj] = (_Float16)acc;
  }
}

// ---------------------------------------------------------------------------
// Kernel 4: flash attention. Block = 8 waves x 16 rows = 128 Q rows; j-blocks
// of 64, double-buffered async-to-LDS staging overlapped with WMMA compute.
// Row sums accumulate inside WMMA (ones-column trick), row max via permlane16.
// ---------------------------------------------------------------------------
__global__ __launch_bounds__(256) void attn_kernel(
    const _Float16* __restrict__ qsf16, const _Float16* __restrict__ kf16,
    const _Float16* __restrict__ vf16T, float* __restrict__ mid)
{
  __shared__ _Float16 Qs[128 * LS];
  __shared__ _Float16 Ks[2][64 * LS];
  __shared__ _Float16 Vs[2][64 * LS];       // [d][j]
  __shared__ _Float16 Ps[8 * 16 * LS];      // per-wave P tile (16 x 64)

  const int tid  = threadIdx.x;
  const int wave = tid >> 5, lane = tid & 31;
  const int t0   = blockIdx.x * 128;
  const int bh   = blockIdx.y;              // b2*8 + h
  const int h    = bh & 7;
  const size_t b2 = (size_t)(bh >> 3);
  const int hi8 = (lane < 16) ? 0 : 8;      // C fragment: VGPR r -> row r / r+8
  const int n16 = lane & 15;                // C/B column
  const int kb  = (lane < 16) ? 0 : 8;      // B fragment K{kb..kb+7, kb+16..kb+23} per half-wave

  // stage Q tile (128 x 64 halves): 4 async ops per wave
  {
    const _Float16* gq = qsf16 + ((size_t)h * N_SEQ + t0) * 64;
    for (int i = tid; i < 128 * 8; i += 256) {
      int r = i >> 3, c = (i & 7) * 8;
      async_load_b128(&Qs[r * LS + c], gq + (size_t)r * 64 + c);
    }
  }

  // K (j-major) + V^T (d-major) block stager: 4 async ops per wave per buffer
  auto stage_kv = [&](int buf, int j0) {
    const _Float16* gk = kf16 + ((size_t)bh * M_SEQ + j0) * 64;
    for (int i = tid; i < 64 * 8; i += 256) {
      int r = i >> 3, c = (i & 7) * 8;
      async_load_b128(&Ks[buf][r * LS + c], gk + (size_t)r * 64 + c);
    }
    const _Float16* gv = vf16T + (size_t)bh * 64 * M_SEQ + j0;
    for (int i = tid; i < 64 * 8; i += 256) {
      int d = i >> 3, c = (i & 7) * 8;
      async_load_b128(&Vs[buf][d * LS + c], gv + (size_t)d * M_SEQ + c);
    }
  };

  stage_kv(0, 0);
  wait_async4();              // Q's 4 ops (oldest, in-order) have retired
  __syncthreads();

  // A fragments (16x32, ISA layout: lanes<16 hold K{0..7,16..23} of row M=lane)
  v16h a0, a1;
  {
    const int m   = wave * 16 + (lane & 15);
    const int off = (lane < 16) ? 0 : 8;
    H16 u0, u1;
    u0.h[0] = *(const v8h*)&Qs[m * LS + off];
    u0.h[1] = *(const v8h*)&Qs[m * LS + off + 16];
    u1.h[0] = *(const v8h*)&Qs[m * LS + 32 + off];
    u1.h[1] = *(const v8h*)&Qs[m * LS + 32 + off + 16];
    a0 = u0.v; a1 = u1.v;
  }

  // ones-column B operand: column n=0 all ones -> O[4] accumulates row sums
  v16h bone;
  {
    _Float16 one = (n16 == 0) ? (_Float16)1.0f : (_Float16)0.0f;
    #pragma unroll
    for (int e = 0; e < 16; ++e) bone[e] = one;
  }

  v8f O[5] = {};              // O[0..3]: out d-frags; O[4] col0: running row sum
  float mrow[8];
  #pragma unroll
  for (int r = 0; r < 8; ++r) mrow[r] = -1e30f;

  for (int jb = 0; jb < M_SEQ / 64; ++jb) {
    const int cur = jb & 1;
    if (jb + 1 < M_SEQ / 64) {          // prefetch next block into other buffer
      stage_kv(cur ^ 1, (jb + 1) * 64);
      wait_async4();                    // current buffer retired, next in flight
    } else {
      wait_async0();
    }
    __syncthreads();

    const _Float16* Kc = &Ks[cur][0];
    const _Float16* Vc = &Vs[cur][0];

    // ---- S = Q K^T (16 rows x 64 cols), f32 accumulation ----
    v8f s[4];
    #pragma unroll
    for (int nf = 0; nf < 4; ++nf) {
      const int jc = nf * 16 + n16;     // B column = K row j
      H16 b0, b1;
      b0.h[0] = *(const v8h*)&Kc[jc * LS + kb];
      b0.h[1] = *(const v8h*)&Kc[jc * LS + kb + 16];
      b1.h[0] = *(const v8h*)&Kc[jc * LS + 32 + kb];
      b1.h[1] = *(const v8h*)&Kc[jc * LS + 32 + kb + 16];
      v8f c = {};
      c = __builtin_amdgcn_wmma_f32_16x16x32_f16(false, a0, false, b0.v, (short)0, c, false, false);
      c = __builtin_amdgcn_wmma_f32_16x16x32_f16(false, a1, false, b1.v, (short)0, c, false, false);
      s[nf] = c;
    }

    // ---- online softmax: row max via VALU permlane16 butterflies ----
    float alpha[8];
    #pragma unroll
    for (int r = 0; r < 8; ++r) {
      float mx = fmaxf(fmaxf(s[0][r], s[1][r]), fmaxf(s[2][r], s[3][r]));
      mx = rowmax16(mx);
      float mn = fmaxf(mrow[r], mx);
      alpha[r] = __expf(mrow[r] - mn);
      mrow[r]  = mn;
      O[0][r] *= alpha[r]; O[1][r] *= alpha[r]; O[2][r] *= alpha[r];
      O[3][r] *= alpha[r]; O[4][r] *= alpha[r];   // running sums rescale too
    }
    #pragma unroll
    for (int nf = 0; nf < 4; ++nf) {
      #pragma unroll
      for (int r = 0; r < 8; ++r) {
        float p = __expf(s[nf][r] - mrow[r]);
        Ps[(wave * 16 + r + hi8) * LS + nf * 16 + n16] = (_Float16)p;
      }
    }

    // ---- O += P [V | 1]  (P from LDS in A-layout; V^T gives contiguous B) ----
    #pragma unroll
    for (int ks = 0; ks < 2; ++ks) {
      H16 ua;
      const int m   = wave * 16 + n16;
      const int pko = ks * 32 + ((lane < 16) ? 0 : 8);
      ua.h[0] = *(const v8h*)&Ps[m * LS + pko];
      ua.h[1] = *(const v8h*)&Ps[m * LS + pko + 16];
      #pragma unroll
      for (int nf = 0; nf < 4; ++nf) {
        const int dc = nf * 16 + n16;   // B column = output d
        const int jj = ks * 32 + kb;    // B K = j
        H16 ub;
        ub.h[0] = *(const v8h*)&Vc[dc * LS + jj];
        ub.h[1] = *(const v8h*)&Vc[dc * LS + jj + 16];
        O[nf] = __builtin_amdgcn_wmma_f32_16x16x32_f16(false, ua.v, false, ub.v, (short)0, O[nf], false, false);
      }
      O[4] = __builtin_amdgcn_wmma_f32_16x16x32_f16(false, ua.v, false, bone, (short)0, O[4], false, false);
    }
    __syncthreads();                    // all reads of cur buffer done
  }

  // normalize (row sums live in O[4] column 0 -> broadcast) and write
  // mid[b2][t][h*64+d]  (t-major for the final 1x1 conv)
  #pragma unroll
  for (int r = 0; r < 8; ++r) {
    float inv = 1.f / bcast16_lane0(O[4][r]);
    int t = t0 + wave * 16 + r + hi8;
    float* o = mid + (b2 * N_SEQ + t) * DIMC + h * 64;
    o[ 0 + n16] = O[0][r] * inv;
    o[16 + n16] = O[1][r] * inv;
    o[32 + n16] = O[2][r] * inv;
    o[48 + n16] = O[3][r] * inv;
  }
}

// ---------------------------------------------------------------------------
// Kernel 5: out = bo + Wo * 0.5*(mid0 + mid1)   (fused stream mean + 1x1 conv)
// ---------------------------------------------------------------------------
__global__ __launch_bounds__(256) void proj_kernel(
    const float* __restrict__ mid, const float* __restrict__ Wo,
    const float* __restrict__ bo, float* __restrict__ out)
{
  __shared__ float xs[32 * 65];
  __shared__ float ws[64 * 65];
  const int t0  = blockIdx.x * 32;
  const int co0 = blockIdx.y * 64;
  const int tid = threadIdx.x;
  const int oc  = tid & 63, ttg = tid >> 6;     // oc fastest -> coalesced stores
  float acc[8] = {};

  for (int ci0 = 0; ci0 < DIMC; ci0 += 64) {
    if (ci0 + 64 < DIMC) {                      // global_prefetch_b8 of next tiles
      __builtin_prefetch(mid + (size_t)(t0 + (tid >> 3)) * DIMC + ci0 + 64 + (tid & 7) * 8, 0, 3);
      __builtin_prefetch(Wo + (size_t)(co0 + (tid >> 2)) * DIMC + ci0 + 64 + (tid & 3) * 16, 0, 3);
    }
    __syncthreads();
    for (int i = tid; i < 32 * 64; i += 256) {
      int tt = i >> 6, ic = i & 63;
      size_t idx = (size_t)(t0 + tt) * DIMC + ci0 + ic;
      xs[tt * 65 + ic] = 0.5f * (mid[idx] + mid[(size_t)N_SEQ * DIMC + idx]);
    }
    for (int i = tid; i < 64 * 64; i += 256) {
      int o = i >> 6, ic = i & 63;
      ws[o * 65 + ic] = Wo[(size_t)(co0 + o) * DIMC + ci0 + ic];
    }
    __syncthreads();
    #pragma unroll
    for (int u = 0; u < 8; ++u) {
      int tt = ttg * 8 + u;
      float a = 0.f;
      #pragma unroll 4
      for (int ic = 0; ic < 64; ++ic) a += ws[oc * 65 + ic] * xs[tt * 65 + ic];
      acc[u] += a;
    }
  }
  #pragma unroll
  for (int u = 0; u < 8; ++u) {
    int tt = ttg * 8 + u;
    out[(size_t)(t0 + tt) * DIMC + co0 + oc] = acc[u] + bo[co0 + oc];
  }
}

// ---------------------------------------------------------------------------
// Workspace layout (bytes). mid aliases q/sidx/sw, which are dead by kernel 4.
// Total required: ~24 MiB.
// ---------------------------------------------------------------------------
#define WS_QS16  ((size_t)0)            // f16 q*scale [8][4096][64]  : 4 MiB
#define WS_K16   ((size_t)4194304)      // f16 K [16][1024][64]       : 2 MiB
#define WS_V16T  ((size_t)6291456)      // f16 V^T [16][64][1024]     : 2 MiB
#define WS_QF32  ((size_t)8388608)      // f32 q [512][4096]          : 8 MiB
#define WS_SIDX  ((size_t)16777216)     // int [8][1024]
#define WS_SW    ((size_t)16809984)     // f32 [8][1024]
#define WS_MID   ((size_t)8388608)      // f32 mid [2][4096][512]     : 16 MiB (aliases q)

extern "C" void kernel_launch(void* const* d_in, const int* in_sizes, int n_in,
                              void* d_out, int out_size, void* d_ws, size_t ws_size,
                              hipStream_t stream)
{
  (void)in_sizes; (void)n_in; (void)out_size; (void)ws_size;
  const float* x      = (const float*)d_in[0];
  const float* prev_x = (const float*)d_in[1];
  const float* Wq     = (const float*)d_in[2];
  const float* Wk     = (const float*)d_in[3];
  const float* Wv     = (const float*)d_in[4];
  const float* Wo     = (const float*)d_in[5];
  const float* bo     = (const float*)d_in[6];
  const float* Wdw    = (const float*)d_in[7];
  const float* bdw    = (const float*)d_in[8];
  const float* Wp     = (const float*)d_in[9];

  char* ws = (char*)d_ws;
  _Float16* qsf16 = (_Float16*)(ws + WS_QS16);
  _Float16* kf16  = (_Float16*)(ws + WS_K16);
  _Float16* vf16T = (_Float16*)(ws + WS_V16T);
  float*    q     = (float*)(ws + WS_QF32);
  int*      sidx  = (int*)(ws + WS_SIDX);
  float*    sw    = (float*)(ws + WS_SW);
  float*    mid   = (float*)(ws + WS_MID);

  qproj_kernel <<<dim3(8, 128), 256, 0, stream>>>(x, prev_x, Wq, q, qsf16);
  offset_kernel<<<dim3(1024),   256, 0, stream>>>(q, Wdw, bdw, Wp, sidx, sw);
  kv_kernel    <<<dim3(16, 32), 256, 0, stream>>>(x, prev_x, Wk, Wv, sidx, sw, kf16, vf16T);
  attn_kernel  <<<dim3(32, 16), 256, 0, stream>>>(qsf16, kf16, vf16T, mid);
  proj_kernel  <<<dim3(128, 8), 256, 0, stream>>>(mid, Wo, bo, (float*)d_out);
}
